// VishwamAILLM_49151605735442
// MI455X (gfx1250) — matrix operand order repaired and checked by reading, not compiled, not run on hardware
//
#include <hip/hip_runtime.h>
#include <stdint.h>

// ---------------------------------------------------------------------------
// Problem constants (B,S,H,D) = (4, 2048, 16, 64), DM = 1024, 3*H*D = 3072
// ---------------------------------------------------------------------------
#define BATCH   4
#define SEQ     2048
#define NHEAD   16
#define DHEAD   64
#define DMODEL  1024
#define NCOL    3072

typedef __attribute__((ext_vector_type(16))) __bf16 v16bf;
typedef __attribute__((ext_vector_type(8)))  float  v8f;
typedef __attribute__((ext_vector_type(4)))  unsigned int v4u;
typedef __attribute__((ext_vector_type(8)))  int v8i;
typedef __attribute__((ext_vector_type(4)))  int v4i;

#if defined(__has_builtin)
#  if __has_builtin(__builtin_amdgcn_tensor_load_to_lds) && \
      __has_builtin(__builtin_amdgcn_s_wait_tensorcnt)
#    define USE_TDM 1
#  endif
#endif
#ifndef USE_TDM
#  define USE_TDM 0
#endif

union Frag { v16bf v; uint32_t u[8]; unsigned short s[16]; };

__device__ __forceinline__ unsigned short f2bf(float x) {
    union { float f; uint32_t u; } c; c.f = x;
    uint32_t u = c.u;
    uint32_t r = u + 0x7FFFu + ((u >> 16) & 1u);   // round-to-nearest-even
    return (unsigned short)(r >> 16);
}

__device__ __forceinline__ v8f wmma_bf16(const Frag& a, const Frag& b, v8f c) {
    return __builtin_amdgcn_wmma_f32_16x16x32_bf16(
        false, a.v, false, b.v, (short)0, c, false, false);
}

// ---------------------------------------------------------------------------
// Kernel 1: qkv = x @ w_qkv  (bf16 WMMA, f32 accum), fused RoPE on q,k.
// Block = 128 threads (4 waves). Block tile: 64 rows x 64 cols (one head-chunk).
// Wave tile: 16 rows x 64 cols -> 4 accumulators of 16x16.
// Writes Q/K/V bf16 in [B,H,S,D] layout into workspace.
// ---------------------------------------------------------------------------
__global__ void __launch_bounds__(128)
qkv_rope_kernel(const float* __restrict__ x, const float* __restrict__ w,
                unsigned short* __restrict__ Qw, unsigned short* __restrict__ Kw,
                unsigned short* __restrict__ Vw)
{
    __shared__ unsigned short xs[64 * 32];   // [row][k]   bf16
    __shared__ unsigned short wT[64 * 32];   // [col][k]   bf16 (transposed)

    const int tid  = threadIdx.x;
    const int lane = tid & 31;
    const int wv   = tid >> 5;        // wave 0..3 -> row sub-tile
    const int g    = lane >> 4;       // half-wave
    const int ln   = lane & 15;

    const int mt = blockIdx.y;        // 64-row tile index (0..127)
    const int nc = blockIdx.x;        // 64-col chunk (0..47): type*16 + head
    const int rowbase = mt * 64;
    const int colbase = nc * 64;

    v8f acc[4];
    #pragma unroll
    for (int t = 0; t < 4; ++t) acc[t] = (v8f){0,0,0,0,0,0,0,0};

    for (int k0 = 0; k0 < DMODEL; k0 += 32) {
        __syncthreads();
        // Stage x tile 64x32 (coalesced) -> bf16 LDS
        #pragma unroll
        for (int i = 0; i < 16; ++i) {
            int e = i * 128 + tid;            // 0..2047
            int r = e >> 5, k = e & 31;
            xs[e] = f2bf(x[(size_t)(rowbase + r) * DMODEL + k0 + k]);
        }
        // Stage w tile 32x64, transposed into [col][k] (coalesced along col)
        #pragma unroll
        for (int i = 0; i < 16; ++i) {
            int e = i * 128 + tid;
            int k = e >> 6, c = e & 63;
            wT[c * 32 + k] = f2bf(w[(size_t)(k0 + k) * NCOL + colbase + c]);
        }
        // Prefetch next K-step tiles into cache (global_prefetch_b8)
        if (k0 + 32 < DMODEL) {
            {   // x: cover 64 rows of the next 128B-wide tile
                int e = tid * 16;
                int r = e >> 5, k = e & 31;
                __builtin_prefetch(&x[(size_t)(rowbase + r) * DMODEL + k0 + 32 + k], 0, 0);
            }
            {   // w: cover 32 rows x 256B
                int e = tid * 16;
                int k = e >> 6, c = e & 63;
                __builtin_prefetch(&w[(size_t)(k0 + 32 + k) * NCOL + colbase + c], 0, 0);
            }
        }
        __syncthreads();

        // A fragment (16x32 bf16): row = ln, k = 2p + 16g
        Frag a;
        {
            const unsigned short* base = &xs[(wv * 16 + ln) * 32];
            #pragma unroll
            for (int p = 0; p < 8; ++p) {
                int k = 2 * p + 16 * g;
                a.u[p] = *(const uint32_t*)(base + k);
            }
        }
        // 4 col tiles: B fragment (32x16 bf16): n = ln, k = 16g + j
        #pragma unroll
        for (int t = 0; t < 4; ++t) {
            Frag b;
            const unsigned short* base = &wT[(t * 16 + ln) * 32];
            #pragma unroll
            for (int p = 0; p < 8; ++p)
                b.u[p] = *(const uint32_t*)(base + 2 * p + 16 * g);
            acc[t] = wmma_bf16(a, b, acc[t]);
        }
    }

    // RoPE (q,k only). Pair (d, d+16) = (acc[0][v], acc[1][v]) in the same lane.
    const int type = nc >> 4;          // 0=q, 1=k, 2=v
    const int h    = nc & 15;
    if (type < 2) {
        // inv_freq[f] = 10000^(-f/16), f = ln
        const float invf = exp2f(-(float)ln * (13.287712379549449f / 16.0f));
        #pragma unroll
        for (int v = 0; v < 8; ++v) {
            int row  = rowbase + wv * 16 + v + 8 * g;
            int spos = row & (SEQ - 1);
            float ang = (float)spos * invf;
            float c = __cosf(ang), si = __sinf(ang);   // native trans ops
            float av = acc[0][v], bv = acc[1][v];
            acc[0][v] = av * c - bv * si;     // d <  16
            acc[1][v] = bv * c + av * si;     // 16 <= d < 32
        }
    }

    unsigned short* dst = (type == 0) ? Qw : (type == 1) ? Kw : Vw;
    #pragma unroll
    for (int v = 0; v < 8; ++v) {
        int row  = rowbase + wv * 16 + v + 8 * g;
        int b    = row >> 11;
        int spos = row & (SEQ - 1);
        size_t base = (((size_t)b * NHEAD + h) * SEQ + spos) * DHEAD;
        #pragma unroll
        for (int t = 0; t < 4; ++t)
            dst[base + t * 16 + ln] = f2bf(acc[t][v]);
    }
}

// ---------------------------------------------------------------------------
// Kernel 2: flash-style causal attention. Grid (S/64, B*H), block = 128 (4 waves).
// Wave: 16 query rows, full D=64. Key loop: 32 keys/iter (causal block bound).
// K tile staged via Tensor Data Mover (tensor_load_to_lds + s_wait_tensorcnt);
// V tile staged transposed manually (TDM cannot transpose).
// QK^T and PV via bf16 WMMA; online softmax with 16-lane shuffles.
// ---------------------------------------------------------------------------
__global__ void __launch_bounds__(128)
attn_kernel(const unsigned short* __restrict__ Qw, const unsigned short* __restrict__ Kw,
            const unsigned short* __restrict__ Vw, float* __restrict__ out)
{
    __shared__ unsigned short sk [32 * 64];     // K tile  [key][d]
    __shared__ unsigned short svT[64 * 32];     // V tile transposed [d][key]
    __shared__ unsigned short sp [4][16 * 32];  // per-wave P staging [q][key]

    const int tid  = threadIdx.x;
    const int lane = tid & 31;
    const int wv   = tid >> 5;
    const int g    = lane >> 4;
    const int ln   = lane & 15;

    const int bh = blockIdx.y;               // b*16 + h
    const int b  = bh >> 4;
    const int h  = bh & 15;
    const int qt = blockIdx.x;
    const int qbase = qt * 64 + wv * 16;
    const size_t hbase = (size_t)bh * SEQ * DHEAD;

    // Resident Q fragments: A 16x32, d-chunks [0,32) and [32,64)
    Frag aq0, aq1;
    {
        const unsigned short* qrow = Qw + hbase + (size_t)(qbase + ln) * DHEAD;
        #pragma unroll
        for (int p = 0; p < 8; ++p) {
            int k = 2 * p + 16 * g;
            aq0.u[p] = *(const uint32_t*)(qrow + k);
            aq1.u[p] = *(const uint32_t*)(qrow + 32 + k);
        }
    }

    v8f o[4];
    #pragma unroll
    for (int t = 0; t < 4; ++t) o[t] = (v8f){0,0,0,0,0,0,0,0};
    float m[8], l[8];
    #pragma unroll
    for (int v = 0; v < 8; ++v) { m[v] = -1e30f; l[v] = 0.0f; }

    const int   kmax = qt * 64 + 63;                 // block-uniform causal bound
    const float SC   = 0.125f * 1.44269504089f;      // (1/sqrt(64)) * log2(e)

    for (int k0 = 0; k0 <= kmax; k0 += 32) {
        __syncthreads();   // protect sk/svT from previous iteration's readers

#if USE_TDM
        // --- TDM: DMA K tile [32 keys x 64 d] bf16 -> sk (wave 0 issues) -----
        if (wv == 0) {
            const uint64_t gaddr = (uint64_t)(uintptr_t)(Kw + hbase + (size_t)k0 * DHEAD);
            const uint32_t laddr = (uint32_t)(uintptr_t)(void*)sk; // LDS byte offset
            v4u g0;
            g0[0] = 1u;                                        // count=1 (valid D#)
            g0[1] = laddr;                                     // lds_addr
            g0[2] = (unsigned)(gaddr & 0xFFFFFFFFu);           // global_addr[31:0]
            g0[3] = (unsigned)((gaddr >> 32) & 0x01FFFFFFu)    // global_addr[56:32]
                  | (2u << 30);                                // type = 2 ("image")
            v8i g1;
            g1[0] = (int)(1u << 16);                           // data_size=1 (2B), wg_mask=0
            g1[1] = (int)((unsigned)DHEAD << 16);              // tensor_dim0 = 64
            g1[2] = (int)((unsigned)SEQ << 16);                // tensor_dim1 = 2048
            g1[3] = (int)((unsigned)DHEAD << 16);              // tile_dim0 = 64
            g1[4] = 32;                                        // tile_dim1 = 32 keys
            g1[5] = DHEAD;                                     // tensor_dim0_stride = 64
            g1[6] = 0;
            g1[7] = 0;                                         // dim1_stride unused (2D tile)
            v4i g2 = (v4i){0,0,0,0};
            v4i g3 = (v4i){0,0,0,0};
            v8i gz = (v8i){0,0,0,0,0,0,0,0};                   // unused trailing group
            __builtin_amdgcn_tensor_load_to_lds(g0, g1, g2, g3, gz, 0);
        }
#else
        // Fallback: cooperative K staging
        #pragma unroll
        for (int i = 0; i < 16; ++i) {
            int e = i * 128 + tid;
            int key = e >> 6, d = e & 63;
            sk[key * 64 + d] = Kw[hbase + (size_t)(k0 + key) * DHEAD + d];
        }
#endif
        // --- V tile staged transposed: svT[d][key] (all 128 threads) --------
        #pragma unroll
        for (int i = 0; i < 16; ++i) {
            int e = i * 128 + tid;
            int key = e >> 6, d = e & 63;
            svT[d * 32 + key] = Vw[hbase + (size_t)(k0 + key) * DHEAD + d];
        }
#if USE_TDM
        if (wv == 0) __builtin_amdgcn_s_wait_tensorcnt(0);
#endif
        __syncthreads();

        // Scores: two 16(key)-tiles, each accumulated over D with 2 WMMAs
        v8f s[2];
        #pragma unroll
        for (int kt = 0; kt < 2; ++kt) {
            v8f accs = (v8f){0,0,0,0,0,0,0,0};
            Frag b0, b1;
            const unsigned short* kr = &sk[(kt * 16 + ln) * 64];
            #pragma unroll
            for (int p = 0; p < 8; ++p) {
                int k = 2 * p + 16 * g;
                b0.u[p] = *(const uint32_t*)(kr + k);        // d in [0,32)
                b1.u[p] = *(const uint32_t*)(kr + 32 + k);   // d in [32,64)
            }
            accs = wmma_bf16(aq0, b0, accs);
            accs = wmma_bf16(aq1, b1, accs);
            s[kt] = accs;
        }

        // Online softmax (row M = v + 8g lives across 16 lanes of one half-wave)
        float p0[8], p1[8], alpha[8];
        #pragma unroll
        for (int v = 0; v < 8; ++v) {
            int q    = qbase + v + 8 * g;
            int key0 = k0 + ln;
            int key1 = k0 + 16 + ln;
            float s0 = (key0 <= q) ? s[0][v] * SC : -1e30f;
            float s1 = (key1 <= q) ? s[1][v] * SC : -1e30f;
            float mx = fmaxf(s0, s1);
            mx = fmaxf(mx, __shfl_xor(mx, 1, 32));
            mx = fmaxf(mx, __shfl_xor(mx, 2, 32));
            mx = fmaxf(mx, __shfl_xor(mx, 4, 32));
            mx = fmaxf(mx, __shfl_xor(mx, 8, 32));
            float mn = fmaxf(m[v], mx);
            float al = exp2f(m[v] - mn);
            float e0 = exp2f(s0 - mn);
            float e1 = exp2f(s1 - mn);
            float rs = e0 + e1;
            rs += __shfl_xor(rs, 1, 32);
            rs += __shfl_xor(rs, 2, 32);
            rs += __shfl_xor(rs, 4, 32);
            rs += __shfl_xor(rs, 8, 32);
            l[v] = l[v] * al + rs;
            m[v] = mn;
            p0[v] = e0; p1[v] = e1; alpha[v] = al;
        }
        #pragma unroll
        for (int v = 0; v < 8; ++v) {
            float al = alpha[v];
            #pragma unroll
            for (int t = 0; t < 4; ++t) o[t][v] *= al;
        }

        // Transpose P (C layout -> A layout) through per-wave LDS staging
        unsigned short* pw = sp[wv];
        #pragma unroll
        for (int v = 0; v < 8; ++v) {
            int M = v + 8 * g;
            pw[M * 32 + ln     ] = f2bf(p0[v]);
            pw[M * 32 + 16 + ln] = f2bf(p1[v]);
        }
        __syncthreads();   // uniform; also orders wave-local LDS write->read
        Frag ap;
        #pragma unroll
        for (int p = 0; p < 8; ++p) {
            int k = 2 * p + 16 * g;
            ap.u[p] = *(const uint32_t*)(&pw[ln * 32 + k]);
        }

        // PV: B fragment from transposed V (n = d, k = key), 4 d-tiles
        #pragma unroll
        for (int t = 0; t < 4; ++t) {
            Frag bv;
            const unsigned short* vr = &svT[(t * 16 + ln) * 32];
            #pragma unroll
            for (int p = 0; p < 8; ++p)
                bv.u[p] = *(const uint32_t*)(vr + 2 * p + 16 * g);
            o[t] = wmma_bf16(ap, bv, o[t]);
        }
    }

    // Epilogue: divide by running sum, write fp32 [b, s, h*64 + d]
    #pragma unroll
    for (int v = 0; v < 8; ++v) {
        int q = qbase + v + 8 * g;
        float inv = 1.0f / l[v];
        size_t obase = ((size_t)b * SEQ + q) * DMODEL + h * DHEAD;
        #pragma unroll
        for (int t = 0; t < 4; ++t)
            out[obase + t * 16 + ln] = o[t][v] * inv;
    }
}

// ---------------------------------------------------------------------------
// Launch
// ---------------------------------------------------------------------------
extern "C" void kernel_launch(void* const* d_in, const int* in_sizes, int n_in,
                              void* d_out, int out_size, void* d_ws, size_t ws_size,
                              hipStream_t stream) {
    (void)in_sizes; (void)n_in; (void)out_size; (void)ws_size;
    const float* x = (const float*)d_in[0];
    const float* w = (const float*)d_in[1];
    // d_in[2] = mask: causal, computed analytically in-kernel.

    const size_t QN = (size_t)BATCH * NHEAD * SEQ * DHEAD;   // 8,388,608 elems
    unsigned short* Qw = (unsigned short*)d_ws;
    unsigned short* Kw = Qw + QN;
    unsigned short* Vw = Kw + QN;
    float* out = (float*)d_out;

    // Kernel 1: 48 col-chunks x 128 row-tiles, 128 threads
    qkv_rope_kernel<<<dim3(NCOL / 64, (BATCH * SEQ) / 64), 128, 0, stream>>>(
        x, w, Qw, Kw, Vw);

    // Kernel 2: (S/64) x (B*H), 128 threads
    attn_kernel<<<dim3(SEQ / 64, BATCH * NHEAD), 128, 0, stream>>>(
        Qw, Kw, Vw, out);
}
